// MultiheadAttention_24103356465278
// MI455X (gfx1250) — matrix operand order repaired and checked by reading, not compiled, not run on hardware
//
#include <hip/hip_runtime.h>
#include <hip/hip_bf16.h>

typedef __attribute__((ext_vector_type(16))) _Float16 v16h;
typedef __attribute__((ext_vector_type(8)))  _Float16 v8h;
typedef __attribute__((ext_vector_type(8)))  float    v8f;
typedef __attribute__((ext_vector_type(4)))  int      v4i;

#define B_ 4
#define T_ 2048
#define C_ 1024
#define H_ 16
#define D_ 64

union V16 { v16h v; v8h h[2]; };

__device__ inline v8f wmma_f16(v16h a, v16h b, v8f c) {
  return __builtin_amdgcn_wmma_f32_16x16x32_f16(false, a, false, b, (short)0, c, false, false);
}

// ---- async global->LDS copy (CDNA5), with safe fallback ----
__device__ inline void async_copy16B(const _Float16* g, _Float16* l) {
#if __has_builtin(__builtin_amdgcn_global_load_async_to_lds_b128)
  __builtin_amdgcn_global_load_async_to_lds_b128(
      (__attribute__((address_space(1))) v4i*)g,
      (__attribute__((address_space(3))) v4i*)l, 0, 0);
#else
  *(v8h*)l = *(const v8h*)g;
#endif
}

template <int N>
__device__ inline void wait_async() {
#if __has_builtin(__builtin_amdgcn_s_wait_asynccnt)
  __builtin_amdgcn_s_wait_asynccnt(N);
#else
  asm volatile("s_wait_asynccnt %0" :: "i"(N) : "memory");
#endif
}

__device__ inline float redmax16(float v) {
  v = fmaxf(v, __shfl_xor(v, 1, 32));
  v = fmaxf(v, __shfl_xor(v, 2, 32));
  v = fmaxf(v, __shfl_xor(v, 4, 32));
  v = fmaxf(v, __shfl_xor(v, 8, 32));
  return v;
}
__device__ inline float redsum16(float v) {
  v += __shfl_xor(v, 1, 32);
  v += __shfl_xor(v, 2, 32);
  v += __shfl_xor(v, 4, 32);
  v += __shfl_xor(v, 8, 32);
  return v;
}

// ---------------------------------------------------------------------------
// fp32 -> f16 convert, 8 elements/thread
// ---------------------------------------------------------------------------
__global__ __launch_bounds__(256) void cvt_f16(const float* __restrict__ src,
                                               _Float16* __restrict__ dst) {
  const size_t i = ((size_t)blockIdx.x * 256 + threadIdx.x) * 8;
  const float4 f0 = ((const float4*)(src + i))[0];
  const float4 f1 = ((const float4*)(src + i))[1];
  v8h h;
  h[0] = (_Float16)f0.x; h[1] = (_Float16)f0.y;
  h[2] = (_Float16)f0.z; h[3] = (_Float16)f0.w;
  h[4] = (_Float16)f1.x; h[5] = (_Float16)f1.y;
  h[6] = (_Float16)f1.z; h[7] = (_Float16)f1.w;
  *(v8h*)(dst + i) = h;
}

// ---------------------------------------------------------------------------
// GEMM: out[m,n] = sum_k A[m,k] * W[n,k] + bias[n]   (A, W both f16)
// block tile 128x128, K-tile 64, double-buffered async LDS staging.
// MODE 0: fp32 out [M, C]
// MODE 1: f16 out head-major [B,H,T,64]   (Q, K)
// MODE 2: f16 out transposed  [B,H,64,T]  (V)
// grid = 512 (64 m-tiles x 8 n-tiles), block = 256 (8 waves)
// ---------------------------------------------------------------------------
template <int MODE>
__global__ __launch_bounds__(256) void gemm128(const _Float16* __restrict__ A,
                                               const _Float16* __restrict__ W,
                                               const float* __restrict__ bias,
                                               void* __restrict__ outp) {
  __shared__ _Float16 Al[2][128][72];
  __shared__ _Float16 Bl[2][128][72];

  const int bm = blockIdx.x >> 3;
  const int bn = blockIdx.x & 7;
  const int tid  = threadIdx.x;
  const int wave = tid >> 5;
  const int lane = tid & 31;
  const int wm = wave >> 1;      // 0..3  -> 32 rows each
  const int wn = wave & 1;       // 0..1  -> 64 cols each
  const int l16  = lane & 15;
  const int hsel = lane >> 4;    // 0 | 1
  const int coff = hsel << 3;    // 0 | 8

  const int m0 = bm * 128;
  const int n0 = bn * 128;

  // staging: each thread async-copies 32 halves of A and 32 of B per tile
  const int sr = tid >> 1;
  const int sc = (tid & 1) * 32;

  auto prefetch = [&](int t, int buf) {
    const _Float16* ag = A + (size_t)(m0 + sr) * C_ + t * 64 + sc;
    const _Float16* bg = W + (size_t)(n0 + sr) * C_ + t * 64 + sc;
#pragma unroll
    for (int i = 0; i < 4; ++i) {
      async_copy16B(ag + i * 8, &Al[buf][sr][sc + i * 8]);
      async_copy16B(bg + i * 8, &Bl[buf][sr][sc + i * 8]);
    }
  };

  v8f acc[2][4] = {};

  prefetch(0, 0);
#pragma unroll 2
  for (int t = 0; t < C_ / 64; ++t) {
    const int cur = t & 1;
    __syncthreads();                       // everyone done reading buf cur^1
    if (t + 1 < C_ / 64) {
      prefetch(t + 1, cur ^ 1);
      wait_async<8>();                     // tile t landed (8 pending for t+1)
    } else {
      wait_async<0>();
    }
    __syncthreads();                       // tile t visible to all waves

#pragma unroll
    for (int kk = 0; kk < 64; kk += 32) {
      V16 af[2];
#pragma unroll
      for (int i = 0; i < 2; ++i) {
        const int r = wm * 32 + i * 16 + l16;
        af[i].h[0] = *(const v8h*)&Al[cur][r][kk + coff];
        af[i].h[1] = *(const v8h*)&Al[cur][r][kk + 16 + coff];
      }
      V16 bf[4];
#pragma unroll
      for (int jn = 0; jn < 4; ++jn) {
        const int r = wn * 64 + jn * 16 + l16;
        const int c = kk + coff;
        bf[jn].h[0] = *(const v8h*)&Bl[cur][r][c];
        bf[jn].h[1] = *(const v8h*)&Bl[cur][r][c + 16];
      }
#pragma unroll
      for (int i = 0; i < 2; ++i)
#pragma unroll
        for (int jn = 0; jn < 4; ++jn)
          acc[i][jn] = wmma_f16(af[i].v, bf[jn].v, acc[i][jn]);
    }
  }

  // ---- store ----
#pragma unroll
  for (int i = 0; i < 2; ++i)
#pragma unroll
    for (int jn = 0; jn < 4; ++jn)
#pragma unroll
      for (int j = 0; j < 8; ++j) {
        const int mrow = m0 + wm * 32 + i * 16 + j + (hsel << 3);
        const int ncol = n0 + wn * 64 + jn * 16 + l16;
        const float val = acc[i][jn][j] + bias[ncol];
        if constexpr (MODE == 0) {
          ((float*)outp)[(size_t)mrow * C_ + ncol] = val;
        } else {
          const int bb = mrow >> 11;          // / T_
          const int tt = mrow & (T_ - 1);
          const int h  = ncol >> 6;           // / D_
          const int d  = ncol & (D_ - 1);
          _Float16* op = (_Float16*)outp;
          if constexpr (MODE == 1)
            op[(((size_t)(bb * H_ + h)) * T_ + tt) * D_ + d] = (_Float16)val;
          else
            op[(((size_t)(bb * H_ + h)) * D_ + d) * T_ + tt] = (_Float16)val;
        }
      }
}

// ---------------------------------------------------------------------------
// Flash attention: Qh,Kh [B,H,T,64] f16, Vt [B,H,64,T] f16 -> attn f16 [B*T, C]
// grid = B*H*(T/128), block = 256 (8 waves, 16 query rows per wave)
// K/V tiles (32 keys) double-buffered via async LDS copies.
// ---------------------------------------------------------------------------
__global__ __launch_bounds__(256) void attention_flash(const _Float16* __restrict__ Qh,
                                                       const _Float16* __restrict__ Kh,
                                                       const _Float16* __restrict__ Vt,
                                                       _Float16* __restrict__ attn) {
  __shared__ _Float16 Kl[2][32][72];    // 32 keys x 64 d
  __shared__ _Float16 Vl[2][64][40];    // 64 d x 32 keys
  __shared__ _Float16 Pl[8][16][40];    // per-wave P scratch 16x32

  const int nqb  = T_ / 128;
  const int bh   = blockIdx.x / nqb;
  const int qblk = blockIdx.x % nqb;
  const int tid  = threadIdx.x;
  const int wave = tid >> 5;
  const int lane = tid & 31;
  const int l16  = lane & 15;
  const int hsel = lane >> 4;
  const int coff = hsel << 3;

  const int q0 = qblk * 128 + wave * 16;
  const _Float16* Kbh = Kh + (size_t)bh * T_ * D_;
  const _Float16* Vbh = Vt + (size_t)bh * D_ * T_;

  auto prefetchKV = [&](int kv0, int buf) {
    const int r  = tid >> 3, ck = (tid & 7) * 8;
    async_copy16B(Kbh + (size_t)(kv0 + r) * D_ + ck, &Kl[buf][r][ck]);
    const int d  = tid >> 2, cv = (tid & 3) * 8;
    async_copy16B(Vbh + (size_t)d * T_ + kv0 + cv, &Vl[buf][d][cv]);
  };

  // Q fragments (A-layout, K-dim = d)
  V16 aq[2];
  {
    const _Float16* qp = Qh + ((size_t)bh * T_ + q0 + l16) * D_;
#pragma unroll
    for (int f = 0; f < 2; ++f) {
      aq[f].h[0] = *(const v8h*)(qp + f * 32 + coff);
      aq[f].h[1] = *(const v8h*)(qp + f * 32 + 16 + coff);
    }
  }

  v8f o[4] = {};
  float mrun[8], lrun[8];
#pragma unroll
  for (int j = 0; j < 8; ++j) { mrun[j] = -1e30f; lrun[j] = 0.0f; }
  const float scale = 0.125f;   // 1/sqrt(64)

  prefetchKV(0, 0);
  for (int it = 0; it < T_ / 32; ++it) {
    const int cur = it & 1;
    __syncthreads();
    if (it + 1 < T_ / 32) {
      prefetchKV((it + 1) * 32, cur ^ 1);
      wait_async<2>();
    } else {
      wait_async<0>();
    }
    __syncthreads();

    // ---- load ALL K fragments first (distinct regs -> loads overlap WMMA) --
    V16 bk0[2], bk1[2];
#pragma unroll
    for (int t = 0; t < 2; ++t) {
      const int kr = t * 16 + l16;
      const int c  = coff;
      bk0[t].h[0] = *(const v8h*)&Kl[cur][kr][c];
      bk0[t].h[1] = *(const v8h*)&Kl[cur][kr][c + 16];
      bk1[t].h[0] = *(const v8h*)&Kl[cur][kr][32 + c];
      bk1[t].h[1] = *(const v8h*)&Kl[cur][kr][32 + c + 16];
    }
    // scores: two 16x16 tiles, K-dim = d (64 = 2 chained WMMAs each)
    v8f s[2];
#pragma unroll
    for (int t = 0; t < 2; ++t) {
      v8f z = {};
      z = wmma_f16(aq[0].v, bk0[t].v, z);
      z = wmma_f16(aq[1].v, bk1[t].v, z);
      s[t] = z;
    }

    // ---- hoist V fragments: their LDS latency hides under softmax VALU ----
    V16 vb[4];
#pragma unroll
    for (int dt = 0; dt < 4; ++dt) {
      const int dcol = dt * 16 + l16;
      const int c = coff;
      vb[dt].h[0] = *(const v8h*)&Vl[cur][dcol][c];
      vb[dt].h[1] = *(const v8h*)&Vl[cur][dcol][c + 16];
    }

    // online softmax (row j on lanes 0-15, row j+8 on lanes 16-31)
    float alpha[8], p0[8], p1[8];
#pragma unroll
    for (int j = 0; j < 8; ++j) {
      const float x0 = s[0][j] * scale;
      const float x1 = s[1][j] * scale;
      const float tm = redmax16(fmaxf(x0, x1));
      const float mn = fmaxf(mrun[j], tm);
      alpha[j] = __expf(mrun[j] - mn);
      p0[j] = __expf(x0 - mn);
      p1[j] = __expf(x1 - mn);
      const float rs = redsum16(p0[j] + p1[j]);
      lrun[j] = lrun[j] * alpha[j] + rs;
      mrun[j] = mn;
    }

    // write P (f16) to per-wave LDS, re-layout as A fragment
#pragma unroll
    for (int j = 0; j < 8; ++j) {
      const int row = j + (hsel << 3);
      Pl[wave][row][l16]      = (_Float16)p0[j];
      Pl[wave][row][16 + l16] = (_Float16)p1[j];
    }
    asm volatile("s_wait_dscnt 0" ::: "memory");

#pragma unroll
    for (int dt = 0; dt < 4; ++dt)
#pragma unroll
      for (int j = 0; j < 8; ++j) o[dt][j] *= alpha[j];

    V16 pa;
    pa.h[0] = *(const v8h*)&Pl[wave][l16][coff];
    pa.h[1] = *(const v8h*)&Pl[wave][l16][16 + coff];

#pragma unroll
    for (int dt = 0; dt < 4; ++dt)
      o[dt] = wmma_f16(pa.v, vb[dt].v, o[dt]);
  }

  // finalize and store to attn [B*T, C] (heads concatenated), f16
#pragma unroll
  for (int j = 0; j < 8; ++j) lrun[j] = 1.0f / lrun[j];
  const int b = bh / H_, h = bh % H_;
#pragma unroll
  for (int dt = 0; dt < 4; ++dt)
#pragma unroll
    for (int j = 0; j < 8; ++j) {
      const int mt = q0 + j + (hsel << 3);
      const int nc = h * D_ + dt * 16 + l16;
      attn[(size_t)(b * T_ + mt) * C_ + nc] = (_Float16)(o[dt][j] * lrun[j]);
    }
}

// ---------------------------------------------------------------------------
extern "C" void kernel_launch(void* const* d_in, const int* in_sizes, int n_in,
                              void* d_out, int out_size, void* d_ws, size_t ws_size,
                              hipStream_t stream) {
  const float* querys = (const float*)d_in[0];
  const float* keys   = (const float*)d_in[1];
  const float* values = (const float*)d_in[2];
  const float* Wq = (const float*)d_in[3];
  const float* bq = (const float*)d_in[4];
  const float* Wk = (const float*)d_in[5];
  const float* bk = (const float*)d_in[6];
  const float* Wv = (const float*)d_in[7];
  const float* bv = (const float*)d_in[8];
  const float* Wo = (const float*)d_in[9];
  const float* bo = (const float*)d_in[10];

  _Float16* ws = (_Float16*)d_ws;
  const size_t MSZ = (size_t)B_ * T_ * C_;     // 8,388,608
  const size_t WSZ = (size_t)C_ * C_;          // 1,048,576
  _Float16* Xq  = ws;
  _Float16* Xk  = Xq + MSZ;
  _Float16* Xv  = Xk + MSZ;
  _Float16* Wqh = Xv + MSZ;
  _Float16* Wkh = Wqh + WSZ;
  _Float16* Wvh = Wkh + WSZ;
  _Float16* Woh = Wvh + WSZ;
  _Float16* Qh  = Woh + WSZ;
  _Float16* Kh  = Qh + MSZ;
  _Float16* Vt  = Kh + MSZ;
  _Float16* At  = Vt + MSZ;                    // total 120 MB

  dim3 blk(256);
  const int cb_in = (int)(MSZ / 2048);         // 4096 blocks
  const int cb_w  = (int)(WSZ / 2048);         // 512 blocks
  cvt_f16<<<dim3(cb_in), blk, 0, stream>>>(querys, Xq);
  cvt_f16<<<dim3(cb_in), blk, 0, stream>>>(keys,   Xk);
  cvt_f16<<<dim3(cb_in), blk, 0, stream>>>(values, Xv);
  cvt_f16<<<dim3(cb_w),  blk, 0, stream>>>(Wq, Wqh);
  cvt_f16<<<dim3(cb_w),  blk, 0, stream>>>(Wk, Wkh);
  cvt_f16<<<dim3(cb_w),  blk, 0, stream>>>(Wv, Wvh);
  cvt_f16<<<dim3(cb_w),  blk, 0, stream>>>(Wo, Woh);

  dim3 gg(512);                                // 64 m-tiles * 8 n-tiles
  gemm128<1><<<gg, blk, 0, stream>>>(Xq, Wqh, bq, Qh);
  gemm128<1><<<gg, blk, 0, stream>>>(Xk, Wkh, bk, Kh);
  gemm128<2><<<gg, blk, 0, stream>>>(Xv, Wvh, bv, Vt);
  attention_flash<<<dim3(B_ * H_ * (T_ / 128)), blk, 0, stream>>>(Qh, Kh, Vt, At);
  gemm128<0><<<gg, blk, 0, stream>>>(At, Woh, bo, d_out);
}
